// Attention_2731599200990
// MI455X (gfx1250) — matrix operand order repaired and checked by reading, not compiled, not run on hardware
//
#include <hip/hip_runtime.h>
#include <hip/hip_bf16.h>

// ---------------------------------------------------------------------------
// CDNA5 (gfx1250) attention: f16 WMMA GEMMs + wave-level flash attention.
// ---------------------------------------------------------------------------

typedef _Float16 v8h  __attribute__((ext_vector_type(8)));
typedef _Float16 v16h __attribute__((ext_vector_type(16)));
typedef float    v8f  __attribute__((ext_vector_type(8)));

#define DIM       768
#define NUM_HEADS 12
#define HEAD_DIM  64
#define SEQ       1024
#define BATCH     8
#define ROWS      (BATCH * SEQ)          // 8192
#define QKV_OUT   (3 * DIM)              // 2304
#define SCALE     0.125f                 // 64^-0.5
#define LDSP      72                     // padded LDS leading dim (halves)

// ---- WMMA wrapper: D = A(16x32 f16) * B(32x16 f16) + C(16x16 f32) ----------
__device__ __forceinline__ v8f wmma_f16(v16h a, v16h b, v8f c) {
  return __builtin_amdgcn_wmma_f32_16x16x32_f16(
      /*neg_a=*/false, a, /*neg_b=*/false, b,
      /*c_mod=*/(short)0, c, /*reuse_a=*/false, /*reuse_b=*/false);
}

// ---- A-fragment (16x32, row-major source [M x ld]) -------------------------
// Lane holds row m = lane&15; element e maps to K = kbase + 16*half + e.
__device__ __forceinline__ v16h load_a_frag(const _Float16* __restrict__ base,
                                            int ld, int row, int kbase,
                                            int m, int hf) {
  const _Float16* p = base + (size_t)(row + m) * ld + kbase + hf * 16;
  v8h lo = *(const v8h*)(p);
  v8h hi = *(const v8h*)(p + 8);
  return __builtin_shufflevector(lo, hi, 0,1,2,3,4,5,6,7,8,9,10,11,12,13,14,15);
}

// ---- B-fragment (32x16). Source is row-major [N x ld] holding B^T ----------
// (i.e. B[k][n] = src[n*ld + k]). Lane holds col n = lane&15; K = kbase + 16*half + e.
__device__ __forceinline__ v16h load_b_frag(const _Float16* __restrict__ base,
                                            int ld, int col, int kbase,
                                            int n, int hf) {
  const _Float16* p = base + (size_t)(col + n) * ld + kbase + hf * 16;
  return *(const v16h*)(p);
}

// ---------------------------------------------------------------------------
// Kernel 0: fp32 -> f16 conversion (grid-stride)
// ---------------------------------------------------------------------------
__global__ void cvt_f32_f16(const float* __restrict__ src,
                            _Float16* __restrict__ dst, int n) {
  int i = blockIdx.x * blockDim.x + threadIdx.x;
  int stride = gridDim.x * blockDim.x;
  for (; i < n; i += stride) dst[i] = (_Float16)src[i];
}

// ---------------------------------------------------------------------------
// Kernel 1: QKV GEMM.  out[i,o] = sum_c x[i,c] * W[o,c]
// One wave computes a 32(M) x 64(N) tile (B frags reused across 2 M tiles).
// Scatters into q_s [B,H,N,D] (pre-scaled), k_h [B,H,N,D], v_t [B,H,D,N].
// grid = (256, 9), block = 128 (4 waves).
// ---------------------------------------------------------------------------
__global__ __launch_bounds__(128) void qkv_gemm(
    const _Float16* __restrict__ xh, const _Float16* __restrict__ wh,
    _Float16* __restrict__ qs, _Float16* __restrict__ kh,
    _Float16* __restrict__ vt) {
  const int lane = threadIdx.x & 31;
  const int wave = threadIdx.x >> 5;
  const int m = lane & 15, hf = lane >> 4;
  const int row  = blockIdx.x * 32;                 // M tile (32 rows)
  const int col0 = (blockIdx.y * 4 + wave) * 64;    // N tile (64 wide)

  v8f acc[2][4] = {{v8f{}, v8f{}, v8f{}, v8f{}}, {v8f{}, v8f{}, v8f{}, v8f{}}};
  for (int k = 0; k < DIM; k += 32) {
    if (k + 32 < DIM) {  // hint next K panel toward the WGP (global_prefetch_b8)
      __builtin_prefetch(xh + (size_t)(row + m + 16 * hf) * DIM + k + 32, 0, 3);
      __builtin_prefetch(wh + (size_t)(col0 + m + 16 * hf) * DIM + k + 32, 0, 3);
    }
    v16h a0 = load_a_frag(xh, DIM, row,      k, m, hf);
    v16h a1 = load_a_frag(xh, DIM, row + 16, k, m, hf);
#pragma unroll
    for (int t = 0; t < 4; ++t) {
      v16h b = load_b_frag(wh, DIM, col0 + t * 16, k, m, hf);
      acc[0][t] = wmma_f16(a0, b, acc[0][t]);
      acc[1][t] = wmma_f16(a1, b, acc[1][t]);
    }
  }

  const int which = col0 / DIM;           // 0=q, 1=k, 2=v (uniform per tile)
  const int h     = (col0 % DIM) / HEAD_DIM;
#pragma unroll
  for (int mi = 0; mi < 2; ++mi)
#pragma unroll
    for (int t = 0; t < 4; ++t) {
      const int d = t * 16 + m;
#pragma unroll
      for (int r = 0; r < 8; ++r) {
        const int i = row + mi * 16 + r + 8 * hf;
        const int b = i >> 10, n = i & 1023;
        const size_t bh = (size_t)(b * NUM_HEADS + h);
        if (which == 0) {
          qs[(bh * SEQ + n) * HEAD_DIM + d] = (_Float16)(acc[mi][t][r] * SCALE);
        } else if (which == 1) {
          kh[(bh * SEQ + n) * HEAD_DIM + d] = (_Float16)acc[mi][t][r];
        } else {
          vt[(bh * HEAD_DIM + d) * SEQ + n] = (_Float16)acc[mi][t][r]; // V^T
        }
      }
    }
}

// ---------------------------------------------------------------------------
// Kernel 2: flash attention.  One wave handles 16 query rows of one (b,h).
// grid = (16 qchunks, 96 bh), block = 128 (4 waves).
// att output layout: [B, N, H*D] = [8192, 768] (proj-ready, f16).
// ---------------------------------------------------------------------------
__global__ __launch_bounds__(128) void attn_kernel(
    const _Float16* __restrict__ qs, const _Float16* __restrict__ kh,
    const _Float16* __restrict__ vt, _Float16* __restrict__ att) {
  __shared__ __align__(32) _Float16 lds[4][16 * LDSP];

  const int lane = threadIdx.x & 31;
  const int wave = threadIdx.x >> 5;
  const int m = lane & 15, hf = lane >> 4;
  const int qtile = blockIdx.x * 4 + wave;          // 0..63
  const int bh = blockIdx.y;                        // 0..95
  const int qrow = qtile * 16;

  const _Float16* Q = qs + (size_t)bh * SEQ * HEAD_DIM;
  const _Float16* K = kh + (size_t)bh * SEQ * HEAD_DIM;
  const _Float16* V = vt + (size_t)bh * HEAD_DIM * SEQ;
  _Float16* P = &lds[wave][0];

  // Q fragments for the whole pass (K-dim = 64 -> two 16x32 frags).
  v16h qf0 = load_a_frag(Q, HEAD_DIM, qrow, 0, m, hf);
  v16h qf1 = load_a_frag(Q, HEAD_DIM, qrow, 32, m, hf);

  v8f o[4] = {v8f{}, v8f{}, v8f{}, v8f{}};
  float mrun[8], lrun[8];
#pragma unroll
  for (int r = 0; r < 8; ++r) { mrun[r] = -1e30f; lrun[r] = 0.0f; }

  for (int kc = 0; kc < SEQ; kc += 64) {
    if (kc + 64 < SEQ) {  // prefetch next 64-key K chunk + V columns
      __builtin_prefetch(K + (size_t)(kc + 64) * HEAD_DIM + lane * 128, 0, 3);
      __builtin_prefetch(V + (size_t)lane * SEQ + kc + 64, 0, 3);
      __builtin_prefetch(V + (size_t)(lane + 32) * SEQ + kc + 64, 0, 3);
    }
    // ---- S = (Q*scale) @ K^T over a 64-key chunk: four 16x16 tiles --------
    v8f s[4];
#pragma unroll
    for (int t = 0; t < 4; ++t) {
      v16h b0 = load_b_frag(K, HEAD_DIM, kc + t * 16, 0, m, hf);
      v16h b1 = load_b_frag(K, HEAD_DIM, kc + t * 16, 32, m, hf);
      v8f z = {};
      s[t] = wmma_f16(qf1, b1, wmma_f16(qf0, b0, z));
    }
    // ---- online softmax (rows live at r + 8*half; cols across 16 lanes) ---
#pragma unroll
    for (int r = 0; r < 8; ++r) {
      float cm = fmaxf(fmaxf(s[0][r], s[1][r]), fmaxf(s[2][r], s[3][r]));
      cm = fmaxf(cm, __shfl_xor(cm, 1));
      cm = fmaxf(cm, __shfl_xor(cm, 2));
      cm = fmaxf(cm, __shfl_xor(cm, 4));
      cm = fmaxf(cm, __shfl_xor(cm, 8));
      const float mnew  = fmaxf(mrun[r], cm);
      const float alpha = __expf(mrun[r] - mnew);
      mrun[r] = mnew;
      lrun[r] *= alpha;
#pragma unroll
      for (int t = 0; t < 4; ++t) o[t][r] *= alpha;
      float psum = 0.0f;
#pragma unroll
      for (int t = 0; t < 4; ++t) {
        const float p = __expf(s[t][r] - mnew);
        psum += p;
        P[(r + 8 * hf) * LDSP + t * 16 + m] = (_Float16)p;   // C-layout -> LDS
      }
      psum += __shfl_xor(psum, 1);
      psum += __shfl_xor(psum, 2);
      psum += __shfl_xor(psum, 4);
      psum += __shfl_xor(psum, 8);
      lrun[r] += psum;
    }
    // ---- reload P as A-fragments (transpose via LDS, same wave, in-order) -
    v16h pf0 = load_a_frag(P, LDSP, 0, 0, m, hf);
    v16h pf1 = load_a_frag(P, LDSP, 0, 32, m, hf);
    // ---- O += P @ V  (V pre-transposed: contiguous B-frag loads) ----------
#pragma unroll
    for (int t = 0; t < 4; ++t) {
      v16h b0 = load_b_frag(V, SEQ, t * 16, kc, m, hf);
      v16h b1 = load_b_frag(V, SEQ, t * 16, kc + 32, m, hf);
      o[t] = wmma_f16(pf1, b1, wmma_f16(pf0, b0, o[t]));
    }
  }

  // ---- normalize and write att[b, n, h*64 + d] -----------------------------
  const int b = bh / NUM_HEADS, h = bh % NUM_HEADS;
#pragma unroll
  for (int r = 0; r < 8; ++r) {
    const float inv = 1.0f / lrun[r];
    const int n = qrow + r + 8 * hf;
#pragma unroll
    for (int t = 0; t < 4; ++t) {
      att[((size_t)(b * SEQ + n)) * DIM + h * HEAD_DIM + t * 16 + m] =
          (_Float16)(o[t][r] * inv);
    }
  }
}

// ---------------------------------------------------------------------------
// Kernel 3: projection GEMM + bias.  out[i,o] = sum_c att[i,c]*W[o,c] + b[o]
// One wave computes a 32x64 tile.  grid = (256, 3), block = 128.
// ---------------------------------------------------------------------------
__global__ __launch_bounds__(128) void proj_gemm(
    const _Float16* __restrict__ att, const _Float16* __restrict__ wh,
    const float* __restrict__ bias, float* __restrict__ out) {
  const int lane = threadIdx.x & 31;
  const int wave = threadIdx.x >> 5;
  const int m = lane & 15, hf = lane >> 4;
  const int row  = blockIdx.x * 32;
  const int col0 = (blockIdx.y * 4 + wave) * 64;

  v8f acc[2][4] = {{v8f{}, v8f{}, v8f{}, v8f{}}, {v8f{}, v8f{}, v8f{}, v8f{}}};
  for (int k = 0; k < DIM; k += 32) {
    if (k + 32 < DIM) {
      __builtin_prefetch(att + (size_t)(row + m + 16 * hf) * DIM + k + 32, 0, 3);
      __builtin_prefetch(wh + (size_t)(col0 + m + 16 * hf) * DIM + k + 32, 0, 3);
    }
    v16h a0 = load_a_frag(att, DIM, row,      k, m, hf);
    v16h a1 = load_a_frag(att, DIM, row + 16, k, m, hf);
#pragma unroll
    for (int t = 0; t < 4; ++t) {
      v16h b = load_b_frag(wh, DIM, col0 + t * 16, k, m, hf);
      acc[0][t] = wmma_f16(a0, b, acc[0][t]);
      acc[1][t] = wmma_f16(a1, b, acc[1][t]);
    }
  }
#pragma unroll
  for (int mi = 0; mi < 2; ++mi)
#pragma unroll
    for (int t = 0; t < 4; ++t) {
      const int o = col0 + t * 16 + m;
      const float bo = bias[o];
#pragma unroll
      for (int r = 0; r < 8; ++r) {
        const int i = row + mi * 16 + r + 8 * hf;
        out[(size_t)i * DIM + o] = acc[mi][t][r] + bo;
      }
    }
}

// ---------------------------------------------------------------------------
extern "C" void kernel_launch(void* const* d_in, const int* in_sizes, int n_in,
                              void* d_out, int out_size, void* d_ws,
                              size_t ws_size, hipStream_t stream) {
  const float* x      = (const float*)d_in[0];   // [8,1024,768]
  const float* qkv_w  = (const float*)d_in[1];   // [2304,768]
  const float* proj_w = (const float*)d_in[2];   // [768,768]
  const float* proj_b = (const float*)d_in[3];   // [768]
  float* out = (float*)d_out;

  _Float16* ws = (_Float16*)d_ws;
  size_t off = 0;
  _Float16* xh    = ws + off; off += (size_t)ROWS * DIM;                       // 6.3M
  _Float16* wqkv  = ws + off; off += (size_t)QKV_OUT * DIM;                    // 1.8M
  _Float16* wproj = ws + off; off += (size_t)DIM * DIM;                        // 0.6M
  _Float16* qs    = ws + off; off += (size_t)BATCH * NUM_HEADS * SEQ * HEAD_DIM;
  _Float16* kh    = ws + off; off += (size_t)BATCH * NUM_HEADS * SEQ * HEAD_DIM;
  _Float16* vt    = ws + off; off += (size_t)BATCH * NUM_HEADS * SEQ * HEAD_DIM;
  _Float16* att   = ws + off; off += (size_t)ROWS * DIM;

  cvt_f32_f16<<<dim3(2048), dim3(256), 0, stream>>>(x,      xh,    ROWS * DIM);
  cvt_f32_f16<<<dim3(1024), dim3(256), 0, stream>>>(qkv_w,  wqkv,  QKV_OUT * DIM);
  cvt_f32_f16<<<dim3(512),  dim3(256), 0, stream>>>(proj_w, wproj, DIM * DIM);

  qkv_gemm   <<<dim3(ROWS / 32, QKV_OUT / 256), dim3(128), 0, stream>>>(
      xh, wqkv, qs, kh, vt);
  attn_kernel<<<dim3(SEQ / 64, BATCH * NUM_HEADS), dim3(128), 0, stream>>>(
      qs, kh, vt, att);
  proj_gemm  <<<dim3(ROWS / 32, DIM / 256), dim3(128), 0, stream>>>(
      att, wproj, proj_b, out);
}
